// MultiheadALiBiSelfAttention_30064771072553
// MI455X (gfx1250) — matrix operand order repaired and checked by reading, not compiled, not run on hardware
//
#include <hip/hip_runtime.h>
#include <hip/hip_bf16.h>

// ---------------------------------------------------------------------------
// Multi-head ALiBi self-attention for MI455X (gfx1250), bf16 WMMA pipeline.
//   B=2, S=2048, D=1024, H=16, Dh=64
// ---------------------------------------------------------------------------

typedef __attribute__((ext_vector_type(16))) __bf16      v16bf;
typedef __attribute__((ext_vector_type(8)))  float       v8f;
typedef __attribute__((ext_vector_type(4)))  unsigned    v4u;

union ABfrag { v4u u4[2]; v16bf v; };            // 32 bytes, POD

#define WMMA_BF16(a, b, c) \
  __builtin_amdgcn_wmma_f32_16x16x32_bf16(false, (a), false, (b), (short)0, (c), false, false)

#define BSZ   2
#define SEQ   2048
#define DEMB  1024
#define NHEAD 16
#define DHEAD 64

// ---------------------------------------------------------------------------
// f32 -> bf16 cast
// ---------------------------------------------------------------------------
__global__ __launch_bounds__(256)
void cast_f32_to_bf16(const float* __restrict__ s, __bf16* __restrict__ d, int n) {
  int i = blockIdx.x * 256 + threadIdx.x;
  if (i < n) d[i] = (__bf16)s[i];
}

// ---------------------------------------------------------------------------
// out[r, o] = sum_i X[r, i] * W[o, i] + bias[o]
// X: [4096, 1024] bf16 row-major, W: [1024, 1024] bf16 row-major ([out, in])
// Wave tile: 16(M) x 64(N).  Block: 8 waves -> 32 x 256.  Grid: (128, 4).
// ---------------------------------------------------------------------------
template<bool OUT_F32>
__global__ __launch_bounds__(256)
void gemm_bt_bias(const __bf16* __restrict__ X, const __bf16* __restrict__ W,
                  const float* __restrict__ bias, void* __restrict__ out)
{
  const int lane = threadIdx.x & 31;
  const int wid  = threadIdx.x >> 5;
  const int m    = lane & 15;          // A row / B col / D col index
  const int hi   = lane >> 4;          // lane half
  const int row0 = blockIdx.x * 32 + (wid >> 2) * 16;
  const int col0 = blockIdx.y * 256 + (wid & 3) * 64;

  v8f acc[4] = {};

  const __bf16* xrow = X + (size_t)(row0 + m) * DEMB;
  const __bf16* wr0  = W + (size_t)(col0 + m) * DEMB;        // n == m

  for (int kk = 0; kk < DEMB; kk += 32) {
    ABfrag a;
    a.u4[0] = *(const v4u*)(xrow + kk + hi * 8);
    a.u4[1] = *(const v4u*)(xrow + kk + 16 + hi * 8);
#pragma unroll
    for (int sub = 0; sub < 4; ++sub) {
      const __bf16* wrow = wr0 + (size_t)(sub * 16) * DEMB;
      ABfrag b;
      b.u4[0] = *(const v4u*)(wrow + kk + hi * 8);
      b.u4[1] = *(const v4u*)(wrow + kk + 16 + hi * 8);
      acc[sub] = WMMA_BF16(a.v, b.v, acc[sub]);
    }
  }

#pragma unroll
  for (int sub = 0; sub < 4; ++sub) {
    const int col = col0 + sub * 16 + m;
    const float bv = bias[col];
#pragma unroll
    for (int r = 0; r < 8; ++r) {
      const int row = row0 + hi * 8 + r;
      const float v = acc[sub][r] + bv;
      if (OUT_F32)
        ((float*)out)[(size_t)row * DEMB + col] = v;
      else
        ((__bf16*)out)[(size_t)row * DEMB + col] = (__bf16)v;
    }
  }
}

// ---------------------------------------------------------------------------
// Flash attention with ALiBi: one wave per (b, h, 16-query tile).
// Streams 32 keys/iteration: 4 QK^T WMMAs, online softmax, P through LDS
// transpose into A-fragment layout, 4 PV WMMAs.
// ---------------------------------------------------------------------------
__global__ __launch_bounds__(32)
void attn_flash(const __bf16* __restrict__ Q, const __bf16* __restrict__ Kmat,
                const __bf16* __restrict__ V, const unsigned char* __restrict__ mask,
                __bf16* __restrict__ O)
{
  __shared__ __bf16 pl[16 * 32];       // P tile staging (16 q-rows x 32 keys)

  const int lane = threadIdx.x;
  const int n    = lane & 15;
  const int hi   = lane >> 4;
  const int qt   = blockIdx.x;         // 0..127
  const int h    = blockIdx.y;         // 0..15
  const int b    = blockIdx.z;         // 0..1
  const int row0 = qt * 16;

  const float slope  = exp2f(-0.5f * (float)(h + 1));
  const float rscale = 0.125f;         // 1/sqrt(64)

  // Q A-fragments for the two 32-wide K(=d) steps covering Dh=64
  const __bf16* qbase = Q + ((size_t)(b * SEQ + row0 + n) * DEMB) + h * DHEAD;
  ABfrag qa[2];
#pragma unroll
  for (int s = 0; s < 2; ++s) {
    qa[s].u4[0] = *(const v4u*)(qbase + s * 32 + hi * 8);
    qa[s].u4[1] = *(const v4u*)(qbase + s * 32 + 16 + hi * 8);
  }

  float mrun[8], lrun[8];
#pragma unroll
  for (int r = 0; r < 8; ++r) { mrun[r] = -INFINITY; lrun[r] = 0.0f; }
  v8f oacc[4] = {};

  for (int k0 = 0; k0 < SEQ; k0 += 32) {
    // ---- scores: S[q, k] for two 16-key subtiles -------------------------
    v8f sc[2] = {};
#pragma unroll
    for (int t = 0; t < 2; ++t) {
      const __bf16* kbase = Kmat + ((size_t)(b * SEQ + k0 + t * 16 + n) * DEMB) + h * DHEAD;
#pragma unroll
      for (int s = 0; s < 2; ++s) {
        ABfrag kb;
        kb.u4[0] = *(const v4u*)(kbase + s * 32 + hi * 8);
        kb.u4[1] = *(const v4u*)(kbase + s * 32 + 16 + hi * 8);
        sc[t] = WMMA_BF16(qa[s].v, kb.v, sc[t]);
      }
    }

    // ---- ALiBi bias, mask, scale, online softmax -------------------------
#pragma unroll
    for (int r = 0; r < 8; ++r) {
      const int q = row0 + hi * 8 + r;
      float p0, p1;
      {
        const int kk = k0 + n;
        float v = (sc[0][r] - slope * fabsf((float)(q - kk))) * rscale;
        if (!mask[((size_t)b * SEQ + q) * SEQ + kk]) v = -INFINITY;
        p0 = v;
      }
      {
        const int kk = k0 + 16 + n;
        float v = (sc[1][r] - slope * fabsf((float)(q - kk))) * rscale;
        if (!mask[((size_t)b * SEQ + q) * SEQ + kk]) v = -INFINITY;
        p1 = v;
      }
      // row max over the 16 lanes of this half-wave
      float mx = fmaxf(p0, p1);
#pragma unroll
      for (int sh = 1; sh < 16; sh <<= 1) mx = fmaxf(mx, __shfl_xor(mx, sh, 32));
      const float mnew  = fmaxf(mrun[r], mx);
      const float alpha = __expf(mrun[r] - mnew);
      mrun[r] = mnew;
      const float e0 = __expf(p0 - mnew);
      const float e1 = __expf(p1 - mnew);
      float rs = e0 + e1;
#pragma unroll
      for (int sh = 1; sh < 16; sh <<= 1) rs += __shfl_xor(rs, sh, 32);
      lrun[r] = lrun[r] * alpha + rs;
#pragma unroll
      for (int sub = 0; sub < 4; ++sub) oacc[sub][r] *= alpha;
      // stage P (bf16) for the LDS transpose into A-fragment layout
      pl[(hi * 8 + r) * 32 + n]      = (__bf16)e0;
      pl[(hi * 8 + r) * 32 + 16 + n] = (__bf16)e1;
    }
    __syncthreads();

    // ---- P A-fragment (16 q x 32 k) --------------------------------------
    ABfrag pa;
    pa.u4[0] = *(const v4u*)(&pl[n * 32 + hi * 16]);
    pa.u4[1] = *(const v4u*)(&pl[n * 32 + hi * 16 + 8]);
    __syncthreads();

    // ---- O += P @ V (four 16-col d subtiles) -----------------------------
#pragma unroll
    for (int sub = 0; sub < 4; ++sub) {
      union { __bf16 e[16]; v16bf v; } vf;
#pragma unroll
      for (int j = 0; j < 16; ++j)
        vf.e[j] = V[((size_t)(b * SEQ + k0 + hi * 16 + j)) * DEMB + h * DHEAD + sub * 16 + n];
      oacc[sub] = WMMA_BF16(pa.v, vf.v, oacc[sub]);
    }
  }

  // ---- normalize and write head-interleaved output (bf16) ----------------
#pragma unroll
  for (int sub = 0; sub < 4; ++sub)
#pragma unroll
    for (int r = 0; r < 8; ++r) {
      const float v = oacc[sub][r] / lrun[r];
      O[((size_t)(b * SEQ + row0 + hi * 8 + r)) * DEMB + h * DHEAD + sub * 16 + n] = (__bf16)v;
    }
}

// ---------------------------------------------------------------------------
// Host-side orchestration
// ---------------------------------------------------------------------------
extern "C" void kernel_launch(void* const* d_in, const int* in_sizes, int n_in,
                              void* d_out, int out_size, void* d_ws, size_t ws_size,
                              hipStream_t stream) {
  const float*         embed = (const float*)d_in[0];
  const unsigned char* mask  = (const unsigned char*)d_in[1];
  const float* Wq = (const float*)d_in[2];
  const float* bq = (const float*)d_in[3];
  const float* Wk = (const float*)d_in[4];
  const float* bk = (const float*)d_in[5];
  const float* Wv = (const float*)d_in[6];
  const float* bv = (const float*)d_in[7];
  const float* Wo = (const float*)d_in[8];
  const float* bo = (const float*)d_in[9];

  char* ws = (char*)d_ws;
  const size_t MB = 1u << 20;
  __bf16* e_bf  = (__bf16*)(ws + 0 * MB);   // 8 MB: embed bf16  [4096,1024]
  __bf16* wq_bf = (__bf16*)(ws + 8 * MB);   // 2 MB each: weights bf16
  __bf16* wk_bf = (__bf16*)(ws + 10 * MB);
  __bf16* wv_bf = (__bf16*)(ws + 12 * MB);
  __bf16* wo_bf = (__bf16*)(ws + 14 * MB);
  __bf16* q_bf  = (__bf16*)(ws + 16 * MB);  // 8 MB each: Q, K, V, attn-out
  __bf16* k_bf  = (__bf16*)(ws + 24 * MB);
  __bf16* v_bf  = (__bf16*)(ws + 32 * MB);
  __bf16* a_bf  = (__bf16*)(ws + 40 * MB);

  const int NE = BSZ * SEQ * DEMB;          // 4 Mi elements
  const int NW = DEMB * DEMB;               // 1 Mi elements

  cast_f32_to_bf16<<<(NE + 255) / 256, 256, 0, stream>>>(embed, e_bf, NE);
  cast_f32_to_bf16<<<(NW + 255) / 256, 256, 0, stream>>>(Wq, wq_bf, NW);
  cast_f32_to_bf16<<<(NW + 255) / 256, 256, 0, stream>>>(Wk, wk_bf, NW);
  cast_f32_to_bf16<<<(NW + 255) / 256, 256, 0, stream>>>(Wv, wv_bf, NW);
  cast_f32_to_bf16<<<(NW + 255) / 256, 256, 0, stream>>>(Wo, wo_bf, NW);

  dim3 gg((BSZ * SEQ) / 32, DEMB / 256);    // (128, 4)
  gemm_bt_bias<false><<<gg, 256, 0, stream>>>(e_bf, wq_bf, bq, q_bf);
  gemm_bt_bias<false><<<gg, 256, 0, stream>>>(e_bf, wk_bf, bk, k_bf);
  gemm_bt_bias<false><<<gg, 256, 0, stream>>>(e_bf, wv_bf, bv, v_bf);

  dim3 ga(SEQ / 16, NHEAD, BSZ);            // (128, 16, 2)
  attn_flash<<<ga, 32, 0, stream>>>(q_bf, k_bf, v_bf, mask, a_bf);

  gemm_bt_bias<true><<<gg, 256, 0, stream>>>(a_bf, wo_bf, bo, (float*)d_out);
}
